// QuantumAttentionFusion_12292196401666
// MI455X (gfx1250) — matrix operand order repaired and checked by reading, not compiled, not run on hardware
//
#include <hip/hip_runtime.h>
#include <hip/hip_bf16.h>

typedef __bf16 bf16;
typedef __bf16 v16bf __attribute__((ext_vector_type(16)));
typedef __bf16 v8bf  __attribute__((ext_vector_type(8)));
typedef float  v8f   __attribute__((ext_vector_type(8)));

#define LN_EPS 1e-5f

// ---------------------------------------------------------------------------
// CDNA5 async global->LDS copy (ASYNCcnt path), with safe fallback.
// Probe-derived signature: (global int4* src, local int4* dst, i32 off, i32 cpol)
// ---------------------------------------------------------------------------
#if __has_builtin(__builtin_amdgcn_global_load_async_to_lds_b128)
#define HAVE_ASYNC_LDS 1
typedef int i32x4 __attribute__((vector_size(16)));
typedef __attribute__((address_space(1))) i32x4 gbl_i32x4;
typedef __attribute__((address_space(3))) i32x4 lds_i32x4;
__device__ __forceinline__ void cp16_async(bf16* l, const bf16* g) {
  __builtin_amdgcn_global_load_async_to_lds_b128((gbl_i32x4*)g, (lds_i32x4*)l,
                                                 0, 0);
}
__device__ __forceinline__ void async_wait0() {
#if __has_builtin(__builtin_amdgcn_s_wait_asynccnt)
  __builtin_amdgcn_s_wait_asynccnt(0);
#else
  asm volatile("s_wait_asynccnt 0" ::: "memory");
#endif
}
#else
#define HAVE_ASYNC_LDS 0
__device__ __forceinline__ void cp16_async(bf16* l, const bf16* g) {
  *(v8bf*)l = *(const v8bf*)g;
}
__device__ __forceinline__ void async_wait0() {}
#endif

// ---------------------------------------------------------------------------
// WMMA fragment helpers (CDNA5 wave32 layouts, CDNA5 ISA §7.12.2)
// A 16x32 bf16: lane&15 = M row; lanes0-15 hold K=0..7,16..23; lanes16-31 hold
//               K=8..15,24..31 (kh = (lane>>4)*8).
// B 32x16 bf16: lane&15 = N col; lanes0-15 hold K=0..7,16..23; lanes16-31 K=8..15,24..31.
// C/D 16x16 f32: VGPR r, lanes0-15 -> (M=r,N=lane); lanes16-31 -> (M=r+8).
// ---------------------------------------------------------------------------
__device__ __forceinline__ v16bf ld_a_frag(const bf16* base, int ld) {
  const int lane = threadIdx.x & 31;
  const int m  = lane & 15;
  const int kh = (lane >> 4) << 3;          // 0 or 8
  const bf16* p = base + (size_t)m * ld + kh;
  v8bf lo = *(const v8bf*)(p);
  v8bf hi = *(const v8bf*)(p + 16);
  v16bf a;
#pragma unroll
  for (int i = 0; i < 8; ++i) { a[i] = lo[i]; a[i + 8] = hi[i]; }
  return a;
}

__device__ __forceinline__ v16bf ld_b_frag(const bf16* base, int ld) {
  const int lane = threadIdx.x & 31;
  const int n  = lane & 15;
  const int kb = (lane >> 4) << 3;          // 0 or 8
  const bf16* p = base + (size_t)n * ld + kb;
  v8bf lo = *(const v8bf*)(p);
  v8bf hi = *(const v8bf*)(p + 16);
  v16bf b;
#pragma unroll
  for (int i = 0; i < 8; ++i) { b[i] = lo[i]; b[i + 8] = hi[i]; }
  return b;
}

__device__ __forceinline__ v8f wmma_bf16(v16bf a, v16bf b, v8f c) {
  return __builtin_amdgcn_wmma_f32_16x16x32_bf16(false, a, false, b,
                                                 (short)0, c, false, false);
}

// ---------------------------------------------------------------------------
// f32 -> bf16 conversion (grid-stride)
// ---------------------------------------------------------------------------
__global__ void cvt_f32_bf16(const float* __restrict__ in,
                             bf16* __restrict__ out, long n) {
  long i = (long)blockIdx.x * blockDim.x + threadIdx.x;
  long stride = (long)gridDim.x * blockDim.x;
  for (; i < n; i += stride) out[i] = (bf16)in[i];
}

// ---------------------------------------------------------------------------
// Fused GEMM:  C[M,N] = epilogue( A[M,K] @ W[N,K]^T + bias )
// 128x128 workgroup tile, 8 waves each 16(M) x 128(N).
// K step 64, double-buffered LDS filled via async global->LDS loads.
// ---------------------------------------------------------------------------
enum { EPI_BN_RELU = 0, EPI_BN_TANH = 1, EPI_BIAS_BF16 = 2, EPI_BIAS_F32 = 3 };

template <int EPI>
__global__ __launch_bounds__(256)
void gemm_bf16_fused(const bf16* __restrict__ A, const bf16* __restrict__ Bw,
                     const float* __restrict__ bias,
                     const float* __restrict__ bn_g,
                     const float* __restrict__ bn_b,
                     const float* __restrict__ bn_m,
                     const float* __restrict__ bn_v,
                     float out_scale, void* __restrict__ out_p,
                     int M, int N, int K) {
  __shared__ __align__(16) bf16 As[2][128][72];   // 64 + 8 pad
  __shared__ __align__(16) bf16 Bs[2][128][72];
  const int tid  = threadIdx.x;
  const int w    = tid >> 5;
  const int lane = tid & 31;
  const int hi   = lane >> 4;
  const int ln   = lane & 15;
  const int gm0  = blockIdx.y * 128;
  const int gn0  = blockIdx.x * 128;

  v8f acc[8];
#pragma unroll
  for (int t = 0; t < 8; ++t)
    acc[t] = (v8f){0.f, 0.f, 0.f, 0.f, 0.f, 0.f, 0.f, 0.f};

  const int srow = tid >> 1;              // 0..127
  const int scol = (tid & 1) << 5;        // 0 or 32

  auto issue_tile = [&](int k0, int bsel) {
    const bf16* ga = A  + (size_t)(gm0 + srow) * K + k0 + scol;
    const bf16* gb = Bw + (size_t)(gn0 + srow) * K + k0 + scol;
    bf16* la = &As[bsel][srow][scol];
    bf16* lb = &Bs[bsel][srow][scol];
#pragma unroll
    for (int c = 0; c < 32; c += 8) {
      cp16_async(la + c, ga + c);
      cp16_async(lb + c, gb + c);
    }
  };

  issue_tile(0, 0);
  async_wait0();
  __syncthreads();

  int buf = 0;
  for (int k0 = 0; k0 < K; k0 += 64) {
    if (k0 + 64 < K) issue_tile(k0 + 64, buf ^ 1);

    const v16bf a0 = ld_a_frag(&As[buf][w * 16][0], 72);
    const v16bf a1 = ld_a_frag(&As[buf][w * 16][32], 72);
#pragma unroll
    for (int nt = 0; nt < 8; ++nt) {
      acc[nt] = wmma_bf16(a0, ld_b_frag(&Bs[buf][nt * 16][0], 72), acc[nt]);
      acc[nt] = wmma_bf16(a1, ld_b_frag(&Bs[buf][nt * 16][32], 72), acc[nt]);
    }

    async_wait0();       // next tile landed (this wave's share)
    __syncthreads();     // all waves' shares visible
    buf ^= 1;
  }

#pragma unroll
  for (int nt = 0; nt < 8; ++nt) {
    const int col = gn0 + nt * 16 + ln;
    const float bb = bias[col];
    float sc = 1.f, sh = 0.f;
    if (EPI == EPI_BN_RELU || EPI == EPI_BN_TANH) {
      const float rs = rsqrtf(bn_v[col] + LN_EPS);
      sc = rs * bn_g[col];
      sh = bn_b[col] - bn_m[col] * sc;
    }
#pragma unroll
    for (int r = 0; r < 8; ++r) {
      const int row = gm0 + w * 16 + r + hi * 8;
      float v = acc[nt][r] + bb;
      if (EPI == EPI_BN_RELU)      { v = fmaxf(v * sc + sh, 0.f); }
      else if (EPI == EPI_BN_TANH) { v = tanhf(v * sc + sh); }
      else                         { v *= out_scale; }
      if (EPI == EPI_BIAS_F32)
        ((float*)out_p)[(size_t)row * N + col] = v;
      else
        ((bf16*)out_p)[(size_t)row * N + col] = (bf16)v;
    }
  }
}

// ---------------------------------------------------------------------------
// Flash attention: one block per (qblock=64, head, batch); 4 waves x 16 rows.
// q is pre-scaled by 1/sqrt(HD) in its projection epilogue.
// ---------------------------------------------------------------------------
__global__ __launch_bounds__(128)
void attention_kernel(const bf16* __restrict__ q, const bf16* __restrict__ k,
                      const bf16* __restrict__ v, bf16* __restrict__ o) {
  const int S = 1024, D = 1024;
  __shared__ __align__(16) bf16 Ks[64][72];      // [key][hd]
  __shared__ __align__(16) bf16 Vt[64][72];      // [hd][key] (transposed)
  __shared__ __align__(16) bf16 Ps[4][16][72];   // per-wave P scratch
  const int tid  = threadIdx.x;
  const int w    = tid >> 5;
  const int lane = tid & 31;
  const int hi   = lane >> 4;
  const int ln   = lane & 15;
  const int b    = blockIdx.z, h = blockIdx.y;
  const int q0   = blockIdx.x * 64 + w * 16;
  const size_t hoff = (size_t)h * 64;

  const bf16* qbase = q + ((size_t)(b * S + q0)) * D + hoff;
  const v16bf qa0 = ld_a_frag(qbase, D);
  const v16bf qa1 = ld_a_frag(qbase + 32, D);

  v8f oacc[4];
#pragma unroll
  for (int t = 0; t < 4; ++t)
    oacc[t] = (v8f){0.f, 0.f, 0.f, 0.f, 0.f, 0.f, 0.f, 0.f};
  float mrun[8], lrun[8];
#pragma unroll
  for (int r = 0; r < 8; ++r) { mrun[r] = -1e30f; lrun[r] = 0.f; }

  for (int kb = 0; kb < S; kb += 64) {
    // ---- stage K tile and transposed V tile into LDS (128 threads) ----
    for (int idx = tid; idx < 64 * 8; idx += 128) {
      const int key = idx >> 3;
      const int c8  = (idx & 7) << 3;
      const size_t g = ((size_t)(b * S + kb + key)) * D + hoff + c8;
      *(v8bf*)&Ks[key][c8] = *(const v8bf*)(k + g);
      v8bf vv = *(const v8bf*)(v + g);
#pragma unroll
      for (int j = 0; j < 8; ++j) Vt[c8 + j][key] = vv[j];
    }
    __syncthreads();

    // ---- scores S = q . k^T (already includes 1/8 scaling via q) ----
    v8f s[4];
#pragma unroll
    for (int nt = 0; nt < 4; ++nt) {
      v8f z = (v8f){0.f, 0.f, 0.f, 0.f, 0.f, 0.f, 0.f, 0.f};
      z = wmma_bf16(qa0, ld_b_frag(&Ks[nt * 16][0], 72), z);
      s[nt] = wmma_bf16(qa1, ld_b_frag(&Ks[nt * 16][32], 72), z);
    }

    // ---- online softmax (rows live in 16-lane half-groups) ----
#pragma unroll
    for (int r = 0; r < 8; ++r) {
      float mx = fmaxf(fmaxf(s[0][r], s[1][r]), fmaxf(s[2][r], s[3][r]));
#pragma unroll
      for (int off = 1; off <= 8; off <<= 1)
        mx = fmaxf(mx, __shfl_xor(mx, off, 32));
      const float mnew  = fmaxf(mrun[r], mx);
      const float alpha = __expf(mrun[r] - mnew);
      float lsum = 0.f;
#pragma unroll
      for (int nt = 0; nt < 4; ++nt) {
        const float p = __expf(s[nt][r] - mnew);
        s[nt][r] = p;
        lsum += p;
      }
#pragma unroll
      for (int off = 1; off <= 8; off <<= 1)
        lsum += __shfl_xor(lsum, off, 32);
      lrun[r] = lrun[r] * alpha + lsum;
      mrun[r] = mnew;
#pragma unroll
      for (int t2 = 0; t2 < 4; ++t2) oacc[t2][r] *= alpha;
#pragma unroll
      for (int nt = 0; nt < 4; ++nt)
        Ps[w][r + hi * 8][nt * 16 + ln] = (bf16)s[nt][r];
    }
    __syncthreads();

    // ---- O += P @ V ----
    const v16bf pa0 = ld_a_frag(&Ps[w][0][0], 72);
    const v16bf pa1 = ld_a_frag(&Ps[w][0][32], 72);
#pragma unroll
    for (int ot = 0; ot < 4; ++ot) {
      oacc[ot] = wmma_bf16(pa0, ld_b_frag(&Vt[ot * 16][0], 72), oacc[ot]);
      oacc[ot] = wmma_bf16(pa1, ld_b_frag(&Vt[ot * 16][32], 72), oacc[ot]);
    }
    __syncthreads();
  }

  // ---- normalize and store attended (B,S,D) bf16 ----
#pragma unroll
  for (int r = 0; r < 8; ++r) {
    const float inv = 1.f / lrun[r];
    const int row = q0 + r + hi * 8;
#pragma unroll
    for (int ot = 0; ot < 4; ++ot)
      o[((size_t)(b * S + row)) * D + hoff + ot * 16 + ln] =
          (bf16)(oacc[ot][r] * inv);
  }
}

// ---------------------------------------------------------------------------
// out = LayerNorm(x + attn_out) * g + b   — one 256-thread block per row
// ---------------------------------------------------------------------------
__global__ __launch_bounds__(256)
void add_layernorm(const float* __restrict__ x, const float* __restrict__ a,
                   const float* __restrict__ g, const float* __restrict__ be,
                   float* __restrict__ out) {
  const int D = 1024;
  __shared__ float red[16];
  const int row = blockIdx.x;
  const int tid = threadIdx.x;
  float y[4];
  float s = 0.f, ss = 0.f;
#pragma unroll
  for (int i = 0; i < 4; ++i) {
    const int c = tid + i * 256;
    const float t = x[(size_t)row * D + c] + a[(size_t)row * D + c];
    y[i] = t; s += t; ss += t * t;
  }
#pragma unroll
  for (int off = 16; off > 0; off >>= 1) {
    s  += __shfl_xor(s, off, 32);
    ss += __shfl_xor(ss, off, 32);
  }
  if ((tid & 31) == 0) { red[tid >> 5] = s; red[8 + (tid >> 5)] = ss; }
  __syncthreads();
  if (tid < 32) {
    float s2  = (tid < 8) ? red[tid] : 0.f;
    float ss2 = (tid < 8) ? red[8 + tid] : 0.f;
#pragma unroll
    for (int off = 4; off > 0; off >>= 1) {
      s2  += __shfl_xor(s2, off, 32);
      ss2 += __shfl_xor(ss2, off, 32);
    }
    if (tid == 0) { red[0] = s2; red[1] = ss2; }
  }
  __syncthreads();
  const float mu  = red[0] / D;
  const float var = red[1] / D - mu * mu;
  const float rs  = rsqrtf(var + LN_EPS);
#pragma unroll
  for (int i = 0; i < 4; ++i) {
    const int c = tid + i * 256;
    out[(size_t)row * D + c] = (y[i] - mu) * rs * g[c] + be[c];
  }
}

// ---------------------------------------------------------------------------
extern "C" void kernel_launch(void* const* d_in, const int* in_sizes, int n_in,
                              void* d_out, int out_size, void* d_ws,
                              size_t ws_size, hipStream_t stream) {
  (void)in_sizes; (void)n_in; (void)out_size;
  const int Bb = 8, S = 1024, D = 1024, D2 = 2048;
  const int M = Bb * S;                        // 8192

  const float* x   = (const float*)d_in[0];
  const float* W1  = (const float*)d_in[1];
  const float* b1  = (const float*)d_in[2];
  const float* g1  = (const float*)d_in[3];
  const float* be1 = (const float*)d_in[4];
  const float* m1  = (const float*)d_in[5];
  const float* v1  = (const float*)d_in[6];
  const float* W2  = (const float*)d_in[7];
  const float* b2  = (const float*)d_in[8];
  const float* g2  = (const float*)d_in[9];
  const float* be2 = (const float*)d_in[10];
  const float* m2  = (const float*)d_in[11];
  const float* v2  = (const float*)d_in[12];
  const float* Wq  = (const float*)d_in[13];
  const float* bq  = (const float*)d_in[14];
  const float* Wk  = (const float*)d_in[15];
  const float* bk  = (const float*)d_in[16];
  const float* Wv  = (const float*)d_in[17];
  const float* bv  = (const float*)d_in[18];
  const float* Wo  = (const float*)d_in[19];
  const float* bo  = (const float*)d_in[20];
  const float* lng = (const float*)d_in[21];
  const float* lnb = (const float*)d_in[22];
  float* out = (float*)d_out;

  char* ws = (char*)d_ws;
  size_t off = 0;
  auto walloc = [&](size_t bytes) -> void* {
    void* p = ws + off;
    off += (bytes + 255) & ~(size_t)255;
    return p;
  };
  bf16* xbf   = (bf16*)walloc((size_t)M * D * 2);
  bf16* W1bf  = (bf16*)walloc((size_t)D2 * D * 2);
  bf16* W2bf  = (bf16*)walloc((size_t)D * D2 * 2);
  bf16* Wqbf  = (bf16*)walloc((size_t)D * D * 2);
  bf16* Wkbf  = (bf16*)walloc((size_t)D * D * 2);
  bf16* Wvbf  = (bf16*)walloc((size_t)D * D * 2);
  bf16* Wobf  = (bf16*)walloc((size_t)D * D * 2);
  bf16* hbf   = (bf16*)walloc((size_t)M * D2 * 2);
  bf16* xebf  = (bf16*)walloc((size_t)M * D * 2);
  bf16* qbf   = (bf16*)walloc((size_t)M * D * 2);
  bf16* kbf   = (bf16*)walloc((size_t)M * D * 2);
  bf16* vbf   = (bf16*)walloc((size_t)M * D * 2);
  bf16* attbf = (bf16*)walloc((size_t)M * D * 2);
  float* aout = (float*)walloc((size_t)M * D * 4);
  if (off > ws_size) return;                   // workspace too small: no-op

  const dim3 cvtb(256);
  auto cvt_grid = [](long n) {
    long g = (n + 255) / 256;
    return dim3((unsigned)(g > 4096 ? 4096 : g));
  };
  cvt_f32_bf16<<<cvt_grid((long)M * D), cvtb, 0, stream>>>(x,  xbf,  (long)M * D);
  cvt_f32_bf16<<<cvt_grid((long)D2 * D), cvtb, 0, stream>>>(W1, W1bf, (long)D2 * D);
  cvt_f32_bf16<<<cvt_grid((long)D * D2), cvtb, 0, stream>>>(W2, W2bf, (long)D * D2);
  cvt_f32_bf16<<<cvt_grid((long)D * D), cvtb, 0, stream>>>(Wq, Wqbf, (long)D * D);
  cvt_f32_bf16<<<cvt_grid((long)D * D), cvtb, 0, stream>>>(Wk, Wkbf, (long)D * D);
  cvt_f32_bf16<<<cvt_grid((long)D * D), cvtb, 0, stream>>>(Wv, Wvbf, (long)D * D);
  cvt_f32_bf16<<<cvt_grid((long)D * D), cvtb, 0, stream>>>(Wo, Wobf, (long)D * D);

  const dim3 gblk(256);
  // h = relu(BN(x @ W1^T + b1))
  gemm_bf16_fused<EPI_BN_RELU><<<dim3(D2 / 128, M / 128), gblk, 0, stream>>>(
      xbf, W1bf, b1, g1, be1, m1, v1, 1.f, hbf, M, D2, D);
  // xe = tanh(BN(h @ W2^T + b2))
  gemm_bf16_fused<EPI_BN_TANH><<<dim3(D / 128, M / 128), gblk, 0, stream>>>(
      hbf, W2bf, b2, g2, be2, m2, v2, 1.f, xebf, M, D, D2);
  // q/k/v projections; 1/sqrt(64) folded into q epilogue
  gemm_bf16_fused<EPI_BIAS_BF16><<<dim3(D / 128, M / 128), gblk, 0, stream>>>(
      xebf, Wqbf, bq, nullptr, nullptr, nullptr, nullptr, 0.125f, qbf, M, D, D);
  gemm_bf16_fused<EPI_BIAS_BF16><<<dim3(D / 128, M / 128), gblk, 0, stream>>>(
      xebf, Wkbf, bk, nullptr, nullptr, nullptr, nullptr, 1.f, kbf, M, D, D);
  gemm_bf16_fused<EPI_BIAS_BF16><<<dim3(D / 128, M / 128), gblk, 0, stream>>>(
      xebf, Wvbf, bv, nullptr, nullptr, nullptr, nullptr, 1.f, vbf, M, D, D);
  // flash attention
  attention_kernel<<<dim3(S / 64, 16, Bb), dim3(128), 0, stream>>>(
      qbf, kbf, vbf, attbf);
  // out-projection (f32)
  gemm_bf16_fused<EPI_BIAS_F32><<<dim3(D / 128, M / 128), gblk, 0, stream>>>(
      attbf, Wobf, bo, nullptr, nullptr, nullptr, nullptr, 1.f, aout, M, D, D);
  // residual + layernorm
  add_layernorm<<<dim3(M), dim3(256), 0, stream>>>(x, aout, lng, lnb, out);
}
